// MA1_56444460204403
// MI455X (gfx1250) — matrix operand order repaired and checked by reading, not compiled, not run on hardware
//
#include <hip/hip_runtime.h>
#include <hip/hip_bf16.h>
#include <cstdint>

typedef __attribute__((ext_vector_type(16))) __bf16 v16bf;
typedef __attribute__((ext_vector_type(8)))  __bf16 v8bf;
typedef __attribute__((ext_vector_type(8)))  float  v8f;
typedef unsigned int u32x4 __attribute__((ext_vector_type(4)));
typedef int          i32x8 __attribute__((ext_vector_type(8)));

#define TILE_M 128
#define TILE_N 128
#define TILE_K 32
#define NTHREADS 256

// ---------------- CDNA5 data movers ----------------
// Async global->LDS 16B copy (VGLOBAL GLOBAL_LOAD_ASYNC_TO_LDS_B128, ASYNCcnt).
__device__ __forceinline__ void async_copy_b128(unsigned lds_off, const void* gaddr) {
  asm volatile("global_load_async_to_lds_b128 %0, %1, off"
               :: "v"(lds_off), "v"(gaddr) : "memory");
}
__device__ __forceinline__ void wait_async0() {
  asm volatile("s_wait_asynccnt 0x0" ::: "memory");
}

// Tensor Data Mover: load a [TILE_M rows x TILE_K elems] bf16 2D tile to LDS.
// D# per CDNA5 ISA (2 SGPR groups, <=2D tensor).
__device__ __forceinline__ void tdm_load_2d(unsigned lds_off, const __bf16* gtile,
                                            int rowStrideElems) {
  const unsigned long long ga = (unsigned long long)(uintptr_t)gtile;
  u32x4 g0;
  g0[0] = 1u;                                   // count=1, user descriptor
  g0[1] = lds_off;                              // LDS byte address
  g0[2] = (unsigned)ga;                         // global addr lo
  g0[3] = (unsigned)(ga >> 32) | 0x80000000u;   // global addr hi | type=2 (image)
  i32x8 g1;
  g1[0] = 0x00010000;                           // data_size=1 (2 bytes)
  g1[1] = 0;                                    // tensor_dim0 = 1<<20 (lo16=0)
  g1[2] = 16;                                   // tensor_dim0 hi=16, tensor_dim1 lo16=0
  g1[3] = 16 | (TILE_K << 16);                  // tensor_dim1 hi=16, tile_dim0=32
  g1[4] = TILE_M;                               // tile_dim1=128, tile_dim2=0
  g1[5] = rowStrideElems;                       // tensor_dim0_stride lo32
  g1[6] = 0;
  g1[7] = 0;
  asm volatile("tensor_load_to_lds %0, %1" :: "s"(g0), "s"(g1) : "memory");
}

// ---------------- WMMA fragment loaders (wave32 layouts, §7.12.2) ----------------
__device__ __forceinline__ v16bf frag_a(const __bf16* base, int lane) {
  const int row = lane & 15, half = lane >> 4;
  const __bf16* p = base + row * TILE_K;
  v8bf lo = *(const v8bf*)(p + half * 8);
  v8bf hi = *(const v8bf*)(p + 16 + half * 8);
  v16bf f;
#pragma unroll
  for (int i = 0; i < 8; ++i) { f[i] = lo[i]; f[i + 8] = hi[i]; }
  return f;
}
__device__ __forceinline__ v16bf frag_b(const __bf16* base, int lane) {
  const int col = lane & 15, half = lane >> 4;
  const __bf16* p = base + col * TILE_K + half * 8;
  v8bf lo = *(const v8bf*)(p);
  v8bf hi = *(const v8bf*)(p + 16);
  v16bf f;
#pragma unroll
  for (int i = 0; i < 8; ++i) { f[i] = lo[i]; f[i + 8] = hi[i]; }
  return f;
}
__device__ __forceinline__ v8f bf_wmma(v16bf a, v16bf b, v8f c) {
  return __builtin_amdgcn_wmma_f32_16x16x32_bf16(false, a, false, b, (short)0, c,
                                                 false, false);
}

// ---------------- GEMM: C = act(scale * A*Bt^T + bias) ----------------
// A bf16 [M][K] row-major; Bt bf16 [N][K] row-major (B pre-transposed).
// A tile staged by TDM (wave 0), B tile by per-lane async copies; double-buffered.
// 8 waves 4x2; each wave 32x64 output -> 8 wmma per 32-deep K step.
template <bool OUT_BF16, bool OUT_TRANS>
__launch_bounds__(NTHREADS)
__global__ void gemm_bf16(const __bf16* __restrict__ Ag, const __bf16* __restrict__ Btg,
                          const float* __restrict__ bias, void* __restrict__ Cvoid,
                          int M, int N, int K, int lda, int ldb, int ldc,
                          long long sA, long long sB, long long sC,
                          float scale, int relu) {
  // [0,8K):As ph0  [8K,16K):As ph1  [16K,24K):Bts ph0  [24K,32K):Bts ph1
  __shared__ alignas(16) char smem[4 * 8192];

  const int tid = threadIdx.x, lane = tid & 31, wave = tid >> 5;
  const int wm = wave >> 1, wn = wave & 1;
  const int rowBase = blockIdx.y * TILE_M, colBase = blockIdx.x * TILE_N;
  const long long zb = blockIdx.z;
  const __bf16* Ab = Ag  + zb * sA;
  const __bf16* Bb = Btg + zb * sB;

  auto issueA = [&](int ph, int k0) {
    if (wave == 0)
      tdm_load_2d((unsigned)(ph * 8192), Ab + (size_t)rowBase * lda + k0, lda);
  };
  auto issueB = [&](int ph, int k0) {
#pragma unroll
    for (int cc = 0; cc < 2; ++cc) {
      const int c = tid + cc * 256;               // 512 x 16B chunks
      const int n = c >> 2, c8 = (c & 3) * 8;
      async_copy_b128((unsigned)(16384 + ph * 8192 + c * 16),
                      Bb + (size_t)(colBase + n) * ldb + k0 + c8);
    }
  };

  v8f acc[2][4] = {};

  issueA(0, 0); issueB(0, 0);
  if (wave == 0) __builtin_amdgcn_s_wait_tensorcnt(0);
  wait_async0();
  __syncthreads();

  int ph = 0;
  for (int k0 = 0; k0 < K; k0 += TILE_K) {
    const bool more = (k0 + TILE_K < K);
    if (more) { issueA(ph ^ 1, k0 + TILE_K); issueB(ph ^ 1, k0 + TILE_K); }
    if (k0 + 2 * TILE_K < K)
      __builtin_prefetch(
          (const char*)(Ab + (size_t)(rowBase + (tid & 127)) * lda + k0 + 2 * TILE_K),
          0, 1);

    const __bf16* Abase = (const __bf16*)(smem + ph * 8192);
    const __bf16* Bbase = (const __bf16*)(smem + 16384 + ph * 8192);
    v16bf a0 = frag_a(Abase + (wm * 32)      * TILE_K, lane);
    v16bf a1 = frag_a(Abase + (wm * 32 + 16) * TILE_K, lane);
#pragma unroll
    for (int ns = 0; ns < 4; ++ns) {
      v16bf b = frag_b(Bbase + (wn * 64 + ns * 16) * TILE_K, lane);
      acc[0][ns] = bf_wmma(a0, b, acc[0][ns]);
      acc[1][ns] = bf_wmma(a1, b, acc[1][ns]);
    }

    if (wave == 0) __builtin_amdgcn_s_wait_tensorcnt(0);
    wait_async0();
    __syncthreads();
    ph ^= 1;
  }

  // ---- epilogue: C/D layout col = lane%16, rows r + 8*(lane/16) ----
  float*  Cf = (float*)Cvoid  + zb * sC;
  __bf16* Cb = (__bf16*)Cvoid + zb * sC;
#pragma unroll
  for (int ms = 0; ms < 2; ++ms) {
#pragma unroll
    for (int ns = 0; ns < 4; ++ns) {
      const int col    = colBase + wn * 64 + ns * 16 + (lane & 15);
      const int rowTop = rowBase + wm * 32 + ms * 16 + (lane >> 4) * 8;
      const float bv = bias ? bias[col] : 0.0f;
      if constexpr (OUT_TRANS) {  // store C^T: 8 consecutive rows -> one 16B store
        union { uint4 u; __bf16 h[8]; } pk;
#pragma unroll
        for (int r = 0; r < 8; ++r) {
          float v = acc[ms][ns][r] * scale + bv;
          if (relu) v = fmaxf(v, 0.0f);
          pk.h[r] = (__bf16)v;
        }
        *(uint4*)(Cb + (size_t)col * ldc + rowTop) = pk.u;
      } else {
#pragma unroll
        for (int r = 0; r < 8; ++r) {
          float v = acc[ms][ns][r] * scale + bv;
          if (relu) v = fmaxf(v, 0.0f);
          const size_t idx = (size_t)(rowTop + r) * ldc + col;
          if (OUT_BF16) Cb[idx] = (__bf16)v; else Cf[idx] = v;
        }
      }
    }
  }
}

// ---------------- prep kernels ----------------
union FV8 { uint4 u; __bf16 h[8]; };

__launch_bounds__(256)
__global__ void cvt_f32_bf16(const float* __restrict__ x, __bf16* __restrict__ y) {
  const long long i = ((long long)blockIdx.x * 256 + threadIdx.x) * 8;
  float4 a = *(const float4*)(x + i);
  float4 b = *(const float4*)(x + i + 4);
  FV8 pk;
  pk.h[0] = (__bf16)a.x; pk.h[1] = (__bf16)a.y; pk.h[2] = (__bf16)a.z; pk.h[3] = (__bf16)a.w;
  pk.h[4] = (__bf16)b.x; pk.h[5] = (__bf16)b.y; pk.h[6] = (__bf16)b.z; pk.h[7] = (__bf16)b.w;
  *(uint4*)(y + i) = pk.u;
}

// W [512][512] f32 -> Wt [512][512] bf16 with Wt[n][k] = W[k][n]
__launch_bounds__(256)
__global__ void transpose_w_bf16(const float* __restrict__ w, __bf16* __restrict__ wt) {
  __shared__ float t[32][33];
  const int tx = threadIdx.x, ty = threadIdx.y;  // 32 x 8
  const int n0 = blockIdx.x * 32, k0 = blockIdx.y * 32;
#pragma unroll
  for (int j = 0; j < 4; ++j)
    t[ty + 8 * j][tx] = w[(size_t)(k0 + ty + 8 * j) * 512 + n0 + tx];
  __syncthreads();
#pragma unroll
  for (int j = 0; j < 4; ++j)
    wt[(size_t)(n0 + ty + 8 * j) * 512 + k0 + tx] = (__bf16)t[tx][ty + 8 * j];
}

// ---------------- softmax (in-place bf16 rows of length 2048) ----------------
__device__ __forceinline__ float wave_max(float v) {
#pragma unroll
  for (int m = 16; m >= 1; m >>= 1) v = fmaxf(v, __shfl_xor(v, m, 32));
  return v;
}
__device__ __forceinline__ float wave_sum(float v) {
#pragma unroll
  for (int m = 16; m >= 1; m >>= 1) v += __shfl_xor(v, m, 32);
  return v;
}
__launch_bounds__(256)
__global__ void softmax_rows(__bf16* __restrict__ P, int L) {
  __shared__ float red[8];
  __bf16* p = P + (size_t)blockIdx.x * L;
  const int t = threadIdx.x;
  float v[8];
  float mx = -3.0e38f;
#pragma unroll
  for (int i = 0; i < 8; ++i) { v[i] = (float)p[t + i * 256]; mx = fmaxf(mx, v[i]); }
  mx = wave_max(mx);
  if ((t & 31) == 0) red[t >> 5] = mx;
  __syncthreads();
  float m = red[0];
#pragma unroll
  for (int i = 1; i < 8; ++i) m = fmaxf(m, red[i]);
  __syncthreads();
  float s = 0.0f;
#pragma unroll
  for (int i = 0; i < 8; ++i) { v[i] = __expf(v[i] - m); s += v[i]; }
  s = wave_sum(s);
  if ((t & 31) == 0) red[t >> 5] = s;
  __syncthreads();
  float tot = 0.0f;
#pragma unroll
  for (int i = 0; i < 8; ++i) tot += red[i];
  const float inv = 1.0f / tot;
#pragma unroll
  for (int i = 0; i < 8; ++i) p[t + i * 256] = (__bf16)(v[i] * inv);
}

// ---------------- driver ----------------
extern "C" void kernel_launch(void* const* d_in, const int* in_sizes, int n_in,
                              void* d_out, int out_size, void* d_ws, size_t ws_size,
                              hipStream_t stream) {
  const float* rep  = (const float*)d_in[0];
  const float* rep1 = (const float*)d_in[1];
  const float* Wq_w = (const float*)d_in[2];
  const float* Wq_b = (const float*)d_in[3];
  const float* Wk_w = (const float*)d_in[4];
  const float* Wk_b = (const float*)d_in[5];
  const float* Wv_w = (const float*)d_in[6];
  const float* Wv_b = (const float*)d_in[7];
  const float* FC_w = (const float*)d_in[8];
  const float* FC_b = (const float*)d_in[9];

  const int BATCH = 8, L = 2048, C = 512;
  const int M = BATCH * L;  // 16384

  // ws layout (bf16): X | X1 | q | k | vT | Wqt Wkt Wvt FCt | P | O
  char* ws = (char*)d_ws;
  const size_t SZ  = (size_t)M * C * sizeof(__bf16);   // 16 MB
  const size_t SW  = (size_t)C * C * sizeof(__bf16);   // 512 KB
  __bf16* X   = (__bf16*)(ws);
  __bf16* X1  = (__bf16*)(ws + SZ);
  __bf16* q   = (__bf16*)(ws + 2 * SZ);
  __bf16* kk  = (__bf16*)(ws + 3 * SZ);
  __bf16* vT  = (__bf16*)(ws + 4 * SZ);
  __bf16* Wqt = (__bf16*)(ws + 5 * SZ);
  __bf16* Wkt = (__bf16*)(ws + 5 * SZ + SW);
  __bf16* Wvt = (__bf16*)(ws + 5 * SZ + 2 * SW);
  __bf16* FCt = (__bf16*)(ws + 5 * SZ + 3 * SW);
  __bf16* P   = (__bf16*)(ws + 5 * SZ + 4 * SW);        // 64 MB
  __bf16* O   = (__bf16*)(ws + 5 * SZ + 4 * SW + (size_t)BATCH * L * L * sizeof(__bf16));

  dim3 blk(NTHREADS);
  const float inv_sqrt_c = 0.044194173824159216f;  // 1/sqrt(512)

  // 0) one-time precision/layout prep
  cvt_f32_bf16<<<dim3((unsigned)((size_t)M * C / 2048)), blk, 0, stream>>>(rep,  X);
  cvt_f32_bf16<<<dim3((unsigned)((size_t)M * C / 2048)), blk, 0, stream>>>(rep1, X1);
  dim3 tb(32, 8), tg(16, 16);
  transpose_w_bf16<<<tg, tb, 0, stream>>>(Wq_w, Wqt);
  transpose_w_bf16<<<tg, tb, 0, stream>>>(Wk_w, Wkt);
  transpose_w_bf16<<<tg, tb, 0, stream>>>(Wv_w, Wvt);
  transpose_w_bf16<<<tg, tb, 0, stream>>>(FC_w, FCt);

  // 1) projections: q,k normal; v written transposed (vT[C][L] per batch)
  dim3 gp(C / TILE_N, M / TILE_M, 1);
  gemm_bf16<true, false><<<gp, blk, 0, stream>>>(
      X,  Wqt, Wq_b, q,  M, C, C, C, C, C, 0, 0, 0, 1.0f, 1);
  gemm_bf16<true, false><<<gp, blk, 0, stream>>>(
      X1, Wkt, Wk_b, kk, M, C, C, C, C, C, 0, 0, 0, 1.0f, 1);
  dim3 gv(C / TILE_N, L / TILE_M, BATCH);
  gemm_bf16<true, true><<<gv, blk, 0, stream>>>(
      X1, Wvt, Wv_b, vT, L, C, C, C, C, L,
      (long long)L * C, 0, (long long)C * L, 1.0f, 1);

  // 2) scores: S = (q . k^T) / sqrt(C)
  dim3 gs(L / TILE_N, L / TILE_M, BATCH);
  gemm_bf16<true, false><<<gs, blk, 0, stream>>>(
      q, kk, nullptr, P, L, L, C, C, C, L,
      (long long)L * C, (long long)L * C, (long long)L * L, inv_sqrt_c, 0);

  // 3) softmax rows, in place
  softmax_rows<<<dim3(M), blk, 0, stream>>>(P, L);

  // 4) O = P . V   (B = vT [C][L] rows)
  dim3 go(C / TILE_N, L / TILE_M, BATCH);
  gemm_bf16<true, false><<<go, blk, 0, stream>>>(
      P, vT, nullptr, O, L, C, L, L, L, C,
      (long long)L * L, (long long)C * L, (long long)L * C, 1.0f, 0);

  // 5) FC: relu(O * FC_w + FC_b) -> f32 output
  gemm_bf16<false, false><<<gp, blk, 0, stream>>>(
      O, FCt, FC_b, d_out, M, C, C, C, C, C, 0, 0, 0, 1.0f, 1);

  (void)in_sizes; (void)n_in; (void)out_size; (void)ws_size;
}
